// NonLocalAttention_4690104287731
// MI455X (gfx1250) — matrix operand order repaired and checked by reading, not compiled, not run on hardware
//
#include <hip/hip_runtime.h>
#include <math.h>

// ---------------- types for WMMA operands ----------------
typedef __attribute__((ext_vector_type(16))) __bf16     v16bf;
typedef __attribute__((ext_vector_type(8)))  float      v8f;
typedef __attribute__((ext_vector_type(4)))  unsigned   v4u;

#define NBATCH 2
#define CCH    64
#define CE     32
#define SPATIAL (96*96)          // 9216

union ABOp { v4u u[2]; unsigned w[8]; unsigned short s[16]; v16bf v; };
union Acc  { v8f v; float f[8]; };

// round-to-nearest-even f32 -> bf16 bits (pure VALU)
__device__ __forceinline__ unsigned short f2bf(float f) {
    unsigned u = __builtin_bit_cast(unsigned, f);
    unsigned r = (u + 0x7FFFu + ((u >> 16) & 1u)) >> 16;
    return (unsigned short)r;
}

// cross-half (lane ^ 16) exchange on the VALU pipe: v_permlanex16_b32.
__device__ __forceinline__ float xhalf_f(float v) {
    int r = __builtin_amdgcn_permlanex16(__builtin_bit_cast(int, v),
                                         __builtin_bit_cast(int, v),
                                         0x76543210, 0xFEDCBA98, false, false);
    return __builtin_bit_cast(float, r);
}
__device__ __forceinline__ unsigned xhalf_u(unsigned v) {
    return (unsigned)__builtin_amdgcn_permlanex16((int)v, (int)v,
                                                  0x76543210, 0xFEDCBA98, false, false);
}

// ---------------------------------------------------------------------------
// Kernel 1: three 1x1-conv + PReLU projections.
//   Q[n][s][0:32]  (bf16 row-major)   -> B-operand (Q^T) friendly
//   K[n][s][0:32]  (bf16 row-major)   -> A-operand friendly
//   VT[n][c][s]    (bf16 transposed)  -> A-operand (V^T) friendly
// ---------------------------------------------------------------------------
__global__ __launch_bounds__(256)
void nla_proj(const float* __restrict__ x,
              const float* __restrict__ w1, const float* __restrict__ b1, const float* __restrict__ a1,
              const float* __restrict__ w2, const float* __restrict__ b2, const float* __restrict__ a2,
              const float* __restrict__ wa, const float* __restrict__ ba, const float* __restrict__ aa,
              unsigned short* __restrict__ Q,
              unsigned short* __restrict__ Km,
              unsigned short* __restrict__ VT)
{
    __shared__ float wsh[2 * CE * CCH + CCH * CCH];   // 8192 floats = 32 KB
    const int tid = threadIdx.x;
    for (int i = tid; i < CE * CCH; i += 256)  wsh[i] = w1[i];
    for (int i = tid; i < CE * CCH; i += 256)  wsh[CE * CCH + i] = w2[i];
    for (int i = tid; i < CCH * CCH; i += 256) wsh[2 * CE * CCH + i] = wa[i];
    __syncthreads();

    const int p = blockIdx.x * 256 + tid;            // 0 .. N*S-1 (72 blocks)
    const int n = p / SPATIAL;
    const int s = p - n * SPATIAL;

    const float* xc = x + (size_t)n * CCH * SPATIAL + s;
    float xv[CCH];
    #pragma unroll
    for (int c = 0; c < CCH; ++c) xv[c] = xc[(size_t)c * SPATIAL];

    const float sl1 = a1[0], sl2 = a2[0], sla = aa[0];

    #pragma unroll 4
    for (int o = 0; o < CE; ++o) {
        float acc = b1[o];
        #pragma unroll
        for (int c = 0; c < CCH; ++c) acc = fmaf(wsh[o * CCH + c], xv[c], acc);
        acc = acc > 0.f ? acc : sl1 * acc;
        Q[((size_t)n * SPATIAL + s) * CE + o] = f2bf(acc);
    }
    #pragma unroll 4
    for (int o = 0; o < CE; ++o) {
        float acc = b2[o];
        #pragma unroll
        for (int c = 0; c < CCH; ++c) acc = fmaf(wsh[CE * CCH + o * CCH + c], xv[c], acc);
        acc = acc > 0.f ? acc : sl2 * acc;
        Km[((size_t)n * SPATIAL + s) * CE + o] = f2bf(acc);
    }
    #pragma unroll 4
    for (int o = 0; o < CCH; ++o) {
        float acc = ba[o];
        #pragma unroll
        for (int c = 0; c < CCH; ++c) acc = fmaf(wsh[2 * CE * CCH + o * CCH + c], xv[c], acc);
        acc = acc > 0.f ? acc : sla * acc;
        VT[((size_t)n * CCH + o) * SPATIAL + s] = f2bf(acc);
    }
}

// ---- load one 32-key block of K (2 A-operands) and V^T (4 A-operands) ----
__device__ __forceinline__ void load_tiles(const unsigned short* __restrict__ Kb,
                                           const unsigned short* __restrict__ Vb,
                                           int jb, int lg, int asel,
                                           ABOp& k0, ABOp& k1, ABOp* va)
{
    const v4u* kp0 = (const v4u*)(Kb + (size_t)(jb + lg) * CE);
    const v4u* kp1 = (const v4u*)(Kb + (size_t)(jb + 16 + lg) * CE);
    k0.u[0] = kp0[2 * asel]; k0.u[1] = kp0[2 * asel + 1];
    k1.u[0] = kp1[2 * asel]; k1.u[1] = kp1[2 * asel + 1];
    #pragma unroll
    for (int t = 0; t < 4; ++t) {
        const v4u* vp = (const v4u*)(Vb + (size_t)(16 * t + lg) * SPATIAL + jb);
        va[t].u[0] = vp[2 * asel];
        va[t].u[1] = vp[2 * asel + 1];
    }
}

// ---- one flash-attention step over a 32-key block (6 WMMAs, zero LDS) ----
__device__ __forceinline__ void attn_step(const ABOp& ka0, const ABOp& ka1, const ABOp* va,
                                          const ABOp& qb, bool hi,
                                          Acc* o, float& mrow, float& lrow)
{
    v8f zero = {};
    // S^T tiles: D[key][query]  (keys in VGPRs, queries across lanes)
    Acc st0, st1;
    st0.v = __builtin_amdgcn_wmma_f32_16x16x32_bf16(false, ka0.v, false, qb.v,
                                                    (short)0, zero, false, false);
    st1.v = __builtin_amdgcn_wmma_f32_16x16x32_bf16(false, ka1.v, false, qb.v,
                                                    (short)0, zero, false, false);

    // online softmax: in-register max tree + one permlanex16 cross-half combine
    float mx = fmaxf(st0.f[0], st1.f[0]);
    #pragma unroll
    for (int r = 1; r < 8; ++r) mx = fmaxf(mx, fmaxf(st0.f[r], st1.f[r]));
    mx = fmaxf(mx, xhalf_f(mx));
    const float mnew  = fmaxf(mrow, mx);
    const float scale = __expf(mrow - mnew);
    mrow = mnew;

    float p0[8], p1[8];
    float ps = 0.f;
    #pragma unroll
    for (int r = 0; r < 8; ++r) {
        p0[r] = __expf(st0.f[r] - mnew);
        p1[r] = __expf(st1.f[r] - mnew);
        ps += p0[r] + p1[r];
    }
    ps += xhalf_f(ps);
    lrow = lrow * scale + ps;
    #pragma unroll
    for (int t = 0; t < 4; ++t)
        #pragma unroll
        for (int r = 0; r < 8; ++r) o[t].f[r] *= scale;

    // assemble P^T B-operand: v_perm_b32 bf16 packs + v_permlanex16 half swap
    unsigned pk0[4], pk1[4], sw0[4], sw1[4];
    #pragma unroll
    for (int v = 0; v < 4; ++v) {
        pk0[v] = __builtin_amdgcn_perm(__builtin_bit_cast(unsigned, p0[2*v+1]),
                                       __builtin_bit_cast(unsigned, p0[2*v]),
                                       0x07060302u);
        pk1[v] = __builtin_amdgcn_perm(__builtin_bit_cast(unsigned, p1[2*v+1]),
                                       __builtin_bit_cast(unsigned, p1[2*v]),
                                       0x07060302u);
        sw0[v] = xhalf_u(pk0[v]);
        sw1[v] = xhalf_u(pk1[v]);
    }
    ABOp pB;
    #pragma unroll
    for (int v = 0; v < 4; ++v) {
        pB.w[v]     = hi ? sw1[v] : pk0[v];   // keys {2v,2v+1}   (+16 for hi half)
        pB.w[4 + v] = hi ? pk1[v] : sw0[v];   // keys {8+2v,9+2v} (+16 for hi half)
    }

    // O^T += V^T x P^T
    #pragma unroll
    for (int t = 0; t < 4; ++t)
        o[t].v = __builtin_amdgcn_wmma_f32_16x16x32_bf16(false, va[t].v, false, pB.v,
                                                         (short)0, o[t].v, false, false);
}

// ---------------------------------------------------------------------------
// Kernel 2: flash attention, transposed formulation, 2-stage software pipeline
// with ping-pong register buffers (prefetch block jb+32 while computing jb).
// ---------------------------------------------------------------------------
__global__ __launch_bounds__(256)
void nla_attn(const unsigned short* __restrict__ Q,
              const unsigned short* __restrict__ Km,
              const unsigned short* __restrict__ VT,
              const float* __restrict__ x,
              float* __restrict__ out)
{
    const int lane = threadIdx.x & 31;
    const int wv   = threadIdx.x >> 5;
    const int lg   = lane & 15;
    const bool hi  = lane >= 16;
    const int asel = hi ? 1 : 0;      // A-operand chunk select
    const int bsel = hi ? 2 : 0;      // B-operand chunk select

    const int blkPerBatch = SPATIAL / 128;               // 72
    const int n  = blockIdx.x / blkPerBatch;
    const int q0 = (blockIdx.x % blkPerBatch) * 128 + wv * 16;

    // Q^T B-operand (one 16-query block, loaded once)
    ABOp qb;
    {
        const v4u* qp = (const v4u*)(Q + ((size_t)n * SPATIAL + q0 + lg) * CE);
        qb.u[0] = qp[bsel];
        qb.u[1] = qp[bsel + 1];
    }

    Acc o[4];
    v8f zero = {};
    #pragma unroll
    for (int t = 0; t < 4; ++t) o[t].v = zero;
    float mrow = -INFINITY;
    float lrow = 0.f;

    const unsigned short* Kb = Km + (size_t)n * SPATIAL * CE;
    const unsigned short* Vb = VT + (size_t)n * CCH * SPATIAL;

    // ---- software pipeline: ping-pong buffers, unrolled x2 (no reg rotation)
    ABOp kA0, kA1, vA[4];        // buffer A
    ABOp kB0, kB1, vB[4];        // buffer B
    load_tiles(Kb, Vb, 0, lg, asel, kA0, kA1, vA);

    for (int jb = 0; jb < SPATIAL; jb += 64) {
        // stage 1: prefetch jb+32 into B, compute block jb from A
        load_tiles(Kb, Vb, jb + 32, lg, asel, kB0, kB1, vB);
        attn_step(kA0, kA1, vA, qb, hi, o, mrow, lrow);

        // stage 2: prefetch jb+64 (wrap at end) into A, compute jb+32 from B
        const int jn = (jb + 64 < SPATIAL) ? jb + 64 : 0;
        load_tiles(Kb, Vb, jn, lg, asel, kA0, kA1, vA);
        attn_step(kB0, kB1, vB, qb, hi, o, mrow, lrow);
    }

    // ---- epilogue: channels in VGPRs, queries across lanes -> coalesced stores
    const float inv = 1.0f / lrow;
    #pragma unroll
    for (int t = 0; t < 4; ++t) {
        #pragma unroll
        for (int r = 0; r < 8; ++r) {
            const int c = 16 * t + (hi ? 8 + r : r);
            const size_t idx = ((size_t)n * CCH + c) * SPATIAL + q0 + lg;
            out[idx] = o[t].f[r] * inv + x[idx];
        }
    }
}

// ---------------------------------------------------------------------------
extern "C" void kernel_launch(void* const* d_in, const int* in_sizes, int n_in,
                              void* d_out, int out_size, void* d_ws, size_t ws_size,
                              hipStream_t stream)
{
    (void)in_sizes; (void)n_in; (void)out_size; (void)ws_size;

    const float* x  = (const float*)d_in[0];
    const float* w1 = (const float*)d_in[1];
    const float* b1 = (const float*)d_in[2];
    const float* a1 = (const float*)d_in[3];
    const float* w2 = (const float*)d_in[4];
    const float* b2 = (const float*)d_in[5];
    const float* a2 = (const float*)d_in[6];
    const float* wa = (const float*)d_in[7];
    const float* ba = (const float*)d_in[8];
    const float* aa = (const float*)d_in[9];
    float* out = (float*)d_out;

    unsigned short* Q  = (unsigned short*)d_ws;                       // N*S*32
    unsigned short* Km = Q  + (size_t)NBATCH * SPATIAL * CE;          // N*S*32
    unsigned short* VT = Km + (size_t)NBATCH * SPATIAL * CE;          // N*64*S

    nla_proj<<<(NBATCH * SPATIAL) / 256, 256, 0, stream>>>(
        x, w1, b1, a1, w2, b2, a2, wa, ba, aa, Q, Km, VT);

    nla_attn<<<(NBATCH * SPATIAL) / 128, 256, 0, stream>>>(Q, Km, VT, x, out);
}
